// CoverageAttention_64467459113464
// MI455X (gfx1250) — matrix operand order repaired and checked by reading, not compiled, not run on hardware
//
#include <hip/hip_runtime.h>
#include <hip/hip_bf16.h>

// ---------------- problem constants (from reference) ----------------
#define B_SZ   32
#define C_DIM  684
#define H_DIM  16
#define W_DIM  48
#define L_DIM  768          // H*W
#define NP_DIM 512
#define OC_DIM 256
#define KTAPS  121          // 11*11
#define KTOT   805          // C_DIM + KTAPS
#define KPAD   832          // 26 * 32 (K=32 WMMA steps)
#define KSTEPS (KPAD / 32)
#define KSPLIT 672          // 21 blocks pure-x; remaining K comes from icol
#define XSTEPS (KSPLIT / 32)        // 21
#define ICOL_ROWS (KPAD - KSPLIT)   // 160: 12 x-chans + 121 taps + 27 zero
#define NTILE  64           // L columns per workgroup
#define NWAVES 16
#define BT_STRIDE 48        // padded LDS row stride (ushorts), keeps 16B align

typedef __attribute__((ext_vector_type(16))) __bf16 v16bf;
typedef __attribute__((ext_vector_type(8)))  float  v8f;

union Frag16 { v16bf v; uint4 q[2]; };

__device__ __forceinline__ unsigned short f2bf(float f) {
  unsigned int u = __float_as_uint(f);
  u += 0x7FFFu + ((u >> 16) & 1u);          // round-to-nearest-even
  return (unsigned short)(u >> 16);
}

#if __has_builtin(__builtin_amdgcn_tanhf)
#define FAST_TANH(v) __builtin_amdgcn_tanhf(v)   // gfx1250 V_TANH_F32
#else
#define FAST_TANH(v) tanhf(v)
#endif

// -------- K0: sum attention history over T=4 -> alpha_sum (B,L) --------
__global__ void alpha_sum_kernel(const float* __restrict__ alpha,
                                 float* __restrict__ asum) {
  int i = blockIdx.x * blockDim.x + threadIdx.x;
  if (i < B_SZ * L_DIM) {
    int b = i / L_DIM, l = i % L_DIM;
    const float* p = alpha + (size_t)b * 4 * L_DIM + l;
    asum[i] = p[0] + p[L_DIM] + p[2 * L_DIM] + p[3 * L_DIM];
  }
}

// -------- K1: bf16 side-panel for K rows 672..831 --------
// icol[b][r][l]: r<12 -> x channel 672+r; 12<=r<133 -> coverage im2col tap
// (zero-padded 11x11); else 0.  Makes every 32-row K-block homogeneous.
__global__ void im2col_kernel(const float* __restrict__ x,
                              const float* __restrict__ asum,
                              unsigned short* __restrict__ icol) {
  int i = blockIdx.x * blockDim.x + threadIdx.x;
  if (i >= B_SZ * ICOL_ROWS * L_DIM) return;
  const int l = i % L_DIM;
  const int t = i / L_DIM;
  const int r = t % ICOL_ROWS;
  const int b = t / ICOL_ROWS;
  float v = 0.f;
  if (r < 12) {                               // tail annotation channels
    v = x[((size_t)(b * C_DIM + KSPLIT + r)) * L_DIM + l];
  } else if (r < 12 + KTAPS) {                // conv taps
    const int kp = r - 12;
    const int dy = kp / 11 - 5, dx = kp % 11 - 5;
    const int yy = l / W_DIM + dy, xx = l % W_DIM + dx;
    if (yy >= 0 && yy < H_DIM && xx >= 0 && xx < W_DIM)
      v = asum[b * L_DIM + yy * W_DIM + xx];
  }
  icol[i] = f2bf(v);
}

// -------- K2: fold conv into GEMM weights --------
// Ucat (bf16, 512 x 832): [U_a | U_f@conv_w | 0-pad]; bf = U_f@conv_b
__global__ __launch_bounds__(128) void build_weights(
    const float* __restrict__ U_a, const float* __restrict__ U_f,
    const float* __restrict__ conv_w, const float* __restrict__ conv_b,
    unsigned short* __restrict__ Ucat, float* __restrict__ bf) {
  const int n = blockIdx.x;                       // 0..511
  for (int k = threadIdx.x; k < KPAD; k += 128) {
    unsigned short o;
    if (k < C_DIM) {
      o = f2bf(U_a[(size_t)n * C_DIM + k]);
    } else if (k < KTOT) {
      const int kp = k - C_DIM;
      float s = 0.f;
      for (int oc = 0; oc < OC_DIM; ++oc)
        s += U_f[(size_t)n * OC_DIM + oc] * conv_w[(size_t)oc * KTAPS + kp];
      o = f2bf(s);
    } else {
      o = 0;
    }
    Ucat[(size_t)n * KPAD + k] = o;
  }
  if (threadIdx.x == 0) {
    float s = 0.f;
    for (int oc = 0; oc < OC_DIM; ++oc)
      s += U_f[(size_t)n * OC_DIM + oc] * conv_b[oc];
    bf[n] = s;
  }
}

// -------- K3: fused (u_a + u_f) GEMM -> tanh -> nu-dot -> e_t --------
// grid (L_DIM/NTILE, B), block 512 (16 waves). Full M=512 per workgroup,
// one 64-column L tile, K in steps of 32 (bf16 WMMA), double-buffered LDS,
// branch-free staging (phase 1: fp32 x, phase 2: bf16 side panel).
__global__ __launch_bounds__(512) void fused_attn_gemm(
    const float* __restrict__ x, const float* __restrict__ u_pred,
    const unsigned short* __restrict__ icol,
    const unsigned short* __restrict__ Ucat,
    const float* __restrict__ bf, const float* __restrict__ nu,
    float* __restrict__ et) {
  __shared__ unsigned short Bt[2][NTILE][BT_STRIDE];  // col-major (col,k)
  __shared__ float part[NWAVES][NTILE];

  const int b    = blockIdx.y;
  const int l0   = blockIdx.x * NTILE;
  const int tid  = threadIdx.x;
  const int wave = tid >> 5;
  const int lane = tid & 31;
  const int lhalf = (lane >> 4) & 1;
  const int l15   = lane & 15;
  const int mbase = wave * 32;                     // 2 M-tiles per wave

  v8f acc[2][4];
  #pragma unroll
  for (int m = 0; m < 2; ++m)
    #pragma unroll
    for (int n = 0; n < 4; ++n) acc[m][n] = (v8f)0.0f;

  // B-tile staging role: this thread fills 4 consecutive cols of one K row
  const int fk = tid >> 4;          // K row within 32-block
  const int fc = (tid & 15) * 4;    // column base within tile

  const float* xrow0 = x + ((size_t)b * C_DIM + fk) * L_DIM + l0 + fc;
  const unsigned short* irow0 = icol + ((size_t)b * ICOL_ROWS + fk) * L_DIM
                                + l0 + fc;

  // one WMMA step from buffer `cur`, weights at K offset k0
  auto compute_step = [&](int k0, int cur) {
    Frag16 afr[2], bfr[4];
    #pragma unroll
    for (int m = 0; m < 2; ++m) {
      const unsigned short* ap =
          Ucat + (size_t)(mbase + m * 16 + l15) * KPAD + k0 + lhalf * 8;
      afr[m].q[0] = *reinterpret_cast<const uint4*>(ap);
      afr[m].q[1] = *reinterpret_cast<const uint4*>(ap + 16);
    }
    #pragma unroll
    for (int n = 0; n < 4; ++n) {
      const unsigned short* bp = &Bt[cur][n * 16 + l15][lhalf * 8];
      bfr[n].q[0] = *reinterpret_cast<const uint4*>(bp);
      bfr[n].q[1] = *reinterpret_cast<const uint4*>(bp + 16);
    }
    if (k0 + 32 < KPAD)
      __builtin_prefetch(Ucat + (size_t)(mbase + l15) * KPAD + k0 + 32, 0, 1);
    #pragma unroll
    for (int m = 0; m < 2; ++m)
      #pragma unroll
      for (int n = 0; n < 4; ++n)
        acc[m][n] = __builtin_amdgcn_wmma_f32_16x16x32_bf16(
            false, afr[m].v, false, bfr[n].v, (short)0, acc[m][n],
            false, false);
  };

  auto store_tile = [&](int buf, ushort4 s) {
    Bt[buf][fc + 0][fk] = s.x; Bt[buf][fc + 1][fk] = s.y;
    Bt[buf][fc + 2][fk] = s.z; Bt[buf][fc + 3][fk] = s.w;
  };

  // ---- prologue: stage K-block 0 (x phase) into buffer 0
  {
    const float4 f4 = *reinterpret_cast<const float4*>(xrow0);
    store_tile(0, {f2bf(f4.x), f2bf(f4.y), f2bf(f4.z), f2bf(f4.w)});
  }
  __syncthreads();

  // ---- phase 1: K-blocks 0..20, staging from fp32 x (branch-free)
  for (int kb = 0; kb < XSTEPS - 1; ++kb) {      // kb = 0..19
    const float4 f4 = *reinterpret_cast<const float4*>(
        xrow0 + (size_t)(kb + 1) * 32 * L_DIM);  // raw load floats to top
    compute_step(kb * 32, kb & 1);
    store_tile((kb & 1) ^ 1,
               {f2bf(f4.x), f2bf(f4.y), f2bf(f4.z), f2bf(f4.w)});
    __syncthreads();
  }
  // ---- kb = 20: compute x block, stage first icol block
  {
    const ushort4 s = *reinterpret_cast<const ushort4*>(irow0);
    compute_step((XSTEPS - 1) * 32, (XSTEPS - 1) & 1);
    store_tile(((XSTEPS - 1) & 1) ^ 1, s);
    __syncthreads();
  }
  // ---- phase 2: K-blocks 21..25, staging from bf16 side panel
  for (int kb = XSTEPS; kb < KSTEPS - 1; ++kb) { // kb = 21..24
    const ushort4 s = *reinterpret_cast<const ushort4*>(
        irow0 + (size_t)(kb + 1 - XSTEPS) * 32 * L_DIM);
    compute_step(kb * 32, kb & 1);
    store_tile((kb & 1) ^ 1, s);
    __syncthreads();
  }
  compute_step((KSTEPS - 1) * 32, (KSTEPS - 1) & 1);  // last block, no stage

  // ---- epilogue: bias + tanh (HW trans op) + nu-weighted reduction over M
  float ps[4] = {0.f, 0.f, 0.f, 0.f};
  #pragma unroll
  for (int m = 0; m < 2; ++m) {
    #pragma unroll
    for (int r = 0; r < 8; ++r) {
      const int mrow = mbase + m * 16 + lhalf * 8 + r;   // D-matrix layout
      const float bias = u_pred[b * NP_DIM + mrow] + bf[mrow];
      const float nuv  = nu[mrow];
      #pragma unroll
      for (int n = 0; n < 4; ++n)
        ps[n] += nuv * FAST_TANH(acc[m][n][r] + bias);
    }
  }
  #pragma unroll
  for (int n = 0; n < 4; ++n) ps[n] += __shfl_xor(ps[n], 16, 32);
  if (lane < 16) {
    #pragma unroll
    for (int n = 0; n < 4; ++n) part[wave][n * 16 + lane] = ps[n];
  }
  __syncthreads();
  if (tid < NTILE) {
    float e = 0.f;
    #pragma unroll
    for (int w = 0; w < NWAVES; ++w) e += part[w][tid];
    et[b * L_DIM + l0 + tid] = e;
  }
}

// -------- K4: softmax over L (in place), one block per batch --------
__global__ __launch_bounds__(256) void softmax_kernel(float* __restrict__ et) {
  const int b = blockIdx.x, t = threadIdx.x;
  __shared__ float red[256];
  float v0 = et[b * L_DIM + t];
  float v1 = et[b * L_DIM + t + 256];
  float v2 = et[b * L_DIM + t + 512];
  red[t] = fmaxf(v0, fmaxf(v1, v2));
  __syncthreads();
  for (int s = 128; s > 0; s >>= 1) {
    if (t < s) red[t] = fmaxf(red[t], red[t + s]);
    __syncthreads();
  }
  const float mx = red[0];
  __syncthreads();
  float e0 = __expf(v0 - mx), e1 = __expf(v1 - mx), e2 = __expf(v2 - mx);
  red[t] = e0 + e1 + e2;
  __syncthreads();
  for (int s = 128; s > 0; s >>= 1) {
    if (t < s) red[t] += red[t + s];
    __syncthreads();
  }
  const float inv = 1.0f / red[0];
  et[b * L_DIM + t]       = e0 * inv;
  et[b * L_DIM + t + 256] = e1 * inv;
  et[b * L_DIM + t + 512] = e2 * inv;
}

// -------- K5: context = alpha_t . a  (memory bound, x read once) --------
__global__ __launch_bounds__(256) void context_kernel(
    const float* __restrict__ alphat, const float* __restrict__ x,
    float* __restrict__ out) {
  const int b = blockIdx.y;
  const int wave = threadIdx.x >> 5, lane = threadIdx.x & 31;
  const int c = blockIdx.x * 8 + wave;
  if (c >= C_DIM) return;
  const float* xp = x + ((size_t)(b * C_DIM + c)) * L_DIM;
  const float* ap = alphat + (size_t)b * L_DIM;
  float s = 0.f;
  for (int l = lane; l < L_DIM; l += 32) s += ap[l] * xp[l];
  #pragma unroll
  for (int d = 16; d > 0; d >>= 1) s += __shfl_xor(s, d, 32);
  if (lane == 0) out[(size_t)b * C_DIM + c] = s;
}

extern "C" void kernel_launch(void* const* d_in, const int* in_sizes, int n_in,
                              void* d_out, int out_size, void* d_ws,
                              size_t ws_size, hipStream_t stream) {
  const float* x      = (const float*)d_in[0];
  const float* u_pred = (const float*)d_in[1];
  const float* alpha  = (const float*)d_in[2];
  const float* conv_w = (const float*)d_in[3];
  const float* conv_b = (const float*)d_in[4];
  const float* U_a    = (const float*)d_in[5];
  const float* U_f    = (const float*)d_in[6];
  const float* nu     = (const float*)d_in[7];
  float* out = (float*)d_out;

  // workspace layout (~8.9 MB, all 16B aligned)
  char* ws = (char*)d_ws;
  size_t off = 0;
  float*          asum = (float*)(ws + off);          off += (size_t)B_SZ * L_DIM * 4;
  unsigned short* Ucat = (unsigned short*)(ws + off); off += (size_t)NP_DIM * KPAD * 2;
  float*          bf   = (float*)(ws + off);          off += (size_t)NP_DIM * 4;
  float*          et   = (float*)(ws + off);          off += (size_t)B_SZ * L_DIM * 4;
  unsigned short* icol = (unsigned short*)(ws + off); // 32*160*768*2 = 7.9 MB

  alpha_sum_kernel<<<(B_SZ * L_DIM + 255) / 256, 256, 0, stream>>>(alpha, asum);
  im2col_kernel<<<(B_SZ * ICOL_ROWS * L_DIM + 255) / 256, 256, 0, stream>>>(
      x, asum, icol);
  build_weights<<<NP_DIM, 128, 0, stream>>>(U_a, U_f, conv_w, conv_b, Ucat, bf);
  fused_attn_gemm<<<dim3(L_DIM / NTILE, B_SZ), 512, 0, stream>>>(
      x, u_pred, icol, Ucat, bf, nu, et);
  softmax_kernel<<<B_SZ, 256, 0, stream>>>(et);
  context_kernel<<<dim3((C_DIM + 7) / 8, B_SZ), 256, 0, stream>>>(et, x, out);
}
